// MultiHeadSelfAttention_50422916055486
// MI455X (gfx1250) — matrix operand order repaired and checked by reading, not compiled, not run on hardware
//
#include <hip/hip_runtime.h>
#include <hip/hip_bf16.h>
#include <math.h>

// ---------------------------------------------------------------------------
// MHA for MI455X (gfx1250, wave32, WMMA).
// All heavy math on v_wmma_f32_16x16x32_f16 (f32 accumulate).
// Shared operands (W panels, K/V tiles) staged into LDS with
// global_load_async_to_lds_b128 (ASYNCcnt), double buffered.
// Scale 1/sqrt(64) * log2(e) folded into Q; softmax done with exp2.
// ---------------------------------------------------------------------------

typedef __attribute__((ext_vector_type(16))) _Float16 v16h;
typedef __attribute__((ext_vector_type(8)))  _Float16 v8h;
typedef __attribute__((ext_vector_type(4)))  _Float16 v4h;
typedef __attribute__((ext_vector_type(8)))  float    v8f;
typedef __attribute__((ext_vector_type(4)))  float    v4f;

#if __has_builtin(__builtin_amdgcn_exp2f)
#define EXP2F(x) __builtin_amdgcn_exp2f(x)
#else
#define EXP2F(x) exp2f(x)
#endif

#define EMBED 1024
#define HEADS 16
#define HD 64
#define SEQ 2048
#define BATCH 2
#define MROWS (BATCH * SEQ)          // 4096
#define QSCALE (0.125f * 1.44269504088896340736f)  // 1/sqrt(64) * log2(e)

__device__ __forceinline__ v8f wmma_f16(v16h a, v16h b, v8f c) {
  // (neg_a, A, neg_b, B, c_mod, C, reuse_a, reuse_b)
  return __builtin_amdgcn_wmma_f32_16x16x32_f16(false, a, false, b, (short)0, c,
                                                false, false);
}

// Async 16-byte copy global -> LDS (per-lane). LDS address = low 32 bits of
// the generic pointer. Tracked by ASYNCcnt.
__device__ __forceinline__ void async_b128(_Float16* lds_dst,
                                           const _Float16* gsrc) {
  unsigned l = (unsigned)(size_t)(void*)lds_dst;
  asm volatile("global_load_async_to_lds_b128 %0, %1, off"
               :: "v"(l), "v"(gsrc) : "memory");
}
__device__ __forceinline__ void wait_async0() {
  asm volatile("s_wait_asynccnt 0" ::: "memory");
}

// A fragment, 16x32 f16 (M x K). Lane l: row = l&15, khalf h = l>>4.
// elements [0..7] = K h*8..h*8+7 ; [8..15] = K 16+h*8..16+h*8+7.
__device__ __forceinline__ v16h load_a_frag(const _Float16* __restrict__ base,
                                            int ld, int lane) {
  const _Float16* p = base + (size_t)(lane & 15) * ld + ((lane >> 4) << 3);
  v8h lo = *(const v8h*)p;
  v8h hi = *(const v8h*)(p + 16);
  v16h a;
#pragma unroll
  for (int i = 0; i < 8; ++i) { a[i] = lo[i]; a[i + 8] = hi[i]; }
  return a;
}

// B fragment, 32x16 f16 (K x N), B[k][n] = src[n*ld + k] ("transposed"
// storage, K contiguous). Lane l: col = l&15, khalf h = l>>4,
// elements [0..7] = K h*8..h*8+7 ; [8..15] = K 16+h*8..16+h*8+7.
// Works for global or LDS-resident src (ld in halves).
__device__ __forceinline__ v16h load_b_frag(const _Float16* base, int ld,
                                            int lane) {
  const _Float16* p = base + (size_t)(lane & 15) * ld + ((lane >> 4) << 3);
  v8h lo = *(const v8h*)p;
  v8h hi = *(const v8h*)(p + 16);
  v16h b;
#pragma unroll
  for (int i = 0; i < 8; ++i) { b[i] = lo[i]; b[i + 8] = hi[i]; }
  return b;
}

// ---------------------------------------------------------------------------
// Kernel 0: f32 -> f16 conversion (vector x4)
// ---------------------------------------------------------------------------
__global__ void cvt_f32_f16_x4(const v4f* __restrict__ in,
                               v4h* __restrict__ out, int n4) {
  int i = blockIdx.x * blockDim.x + threadIdx.x;
  if (i < n4) {
    v4f v = in[i];
    v4h h;
#pragma unroll
    for (int k = 0; k < 4; ++k) h[k] = (_Float16)v[k];
    out[i] = h;
  }
}

// ---------------------------------------------------------------------------
// GEMM building block: block = 4 waves = 64(M) x 64(N) tile; the 64x32 B
// panel slice is async-staged into LDS once per k-step and shared by all 4
// waves (4x cut in W traffic). Double buffered. Layout Bs: [n][32 k halves].
// ---------------------------------------------------------------------------
__device__ __forceinline__ void stage_b_panel(const _Float16* __restrict__ w,
                                              _Float16* dst, int tid) {
  // 64 rows x 64 bytes; 256 16B chunks over 128 threads.
#pragma unroll
  for (int c = tid; c < 256; c += 128) {
    const int row = c >> 2, q = c & 3;
    async_b128(dst + row * 32 + q * 8, w + (size_t)row * EMBED + q * 8);
  }
}

// ---------------------------------------------------------------------------
// Kernel 1: QKV projection.  C[m][n] = sum_k x[m][k]*W[n][k] + b[n]
// grid: (3072/64, 4096/64), block 128 (4 waves).
// Epilogue scatters into Q (scaled), K, V^T (f16).
// ---------------------------------------------------------------------------
__global__ __launch_bounds__(128) void qkv_gemm(
    const _Float16* __restrict__ xh, const _Float16* __restrict__ wh,
    const float* __restrict__ bias, _Float16* __restrict__ Qh,
    _Float16* __restrict__ Kh, _Float16* __restrict__ Vth) {
  __shared__ __align__(32) _Float16 Bs[2][64 * 32];
  const int tid = threadIdx.x;
  const int lane = tid & 31;
  const int wid = tid >> 5;
  const int m0 = blockIdx.y * 64 + wid * 16;
  const int n0 = blockIdx.x * 64;
  const _Float16* wpan = wh + (size_t)n0 * EMBED;

  stage_b_panel(wpan, Bs[0], tid);

  v8f acc[4] = {};
  for (int k = 0; k < EMBED; k += 32) {
    const int buf = (k >> 5) & 1;
    wait_async0();          // my async copies for Bs[buf] have landed
    __syncthreads();        // everyone's copies have landed
    if (k + 32 < EMBED) stage_b_panel(wpan + (k + 32), Bs[buf ^ 1], tid);

    v16h a = load_a_frag(xh + (size_t)m0 * EMBED + k, EMBED, lane);
#pragma unroll
    for (int j = 0; j < 4; ++j) {
      v16h b = load_b_frag(&Bs[buf][j * 16 * 32], 32, lane);
      acc[j] = wmma_f16(a, b, acc[j]);
    }
    __syncthreads();        // done reading Bs[buf] before it is refilled
  }

  const int col = lane & 15;
  const int rh = lane >> 4;   // row half: rows rh*8 .. rh*8+7
#pragma unroll
  for (int j = 0; j < 4; ++j) {
    const int n = n0 + j * 16 + col;
    const float bv = bias[n];
    const int which = n >> 10;      // 0=Q 1=K 2=V (uniform per 16-col tile)
    const int within = n & 1023;
    const int head = within >> 6;   // uniform per tile
    const int d = within & 63;
#pragma unroll
    for (int r = 0; r < 8; ++r) {
      const int m = m0 + rh * 8 + r;
      const int b = m >> 11;        // batch
      const int s = m & 2047;       // seq
      const int bh = b * HEADS + head;
      const float v = acc[j][r] + bv;
      if (which == 0)
        Qh[((size_t)bh * SEQ + s) * HD + d] = (_Float16)(v * QSCALE);
      else if (which == 1)
        Kh[((size_t)bh * SEQ + s) * HD + d] = (_Float16)v;
      else
        Vth[((size_t)bh * HD + d) * SEQ + s] = (_Float16)v;
    }
  }
}

// ---------------------------------------------------------------------------
// Kernel 2: flash attention. Block = 4 waves, one (b,h), 64 queries.
// Per 32-key block: K tile (32x64) and V^T tile (64x32) async-staged into
// LDS once per block and shared by the 4 waves (4x cut in K/V traffic),
// double buffered. 4 WMMA for scores, online softmax (log2 domain), P staged
// through per-wave LDS (C-layout -> A-layout), 4 WMMA for PV.
// grid: (2048/64, B*H), block 128.
// ---------------------------------------------------------------------------
__global__ __launch_bounds__(128) void flash_attn(
    const _Float16* __restrict__ Qh, const _Float16* __restrict__ Kh,
    const _Float16* __restrict__ Vth, _Float16* __restrict__ Oh) {
  __shared__ __align__(32) _Float16 Ks[2][32 * 64];  // [key][hd]
  __shared__ __align__(32) _Float16 Vs[2][64 * 32];  // [hd][key]
  __shared__ __align__(32) _Float16 pbuf[4][16 * 32];

  const int tid = threadIdx.x;
  const int lane = tid & 31;
  const int wid = tid >> 5;
  const int bh = blockIdx.y;                 // 0..31
  const int q0 = blockIdx.x * 64 + wid * 16; // query tile base
  const int col = lane & 15;
  const int rh = lane >> 4;

  const _Float16* Qb = Qh + (size_t)bh * SEQ * HD;
  const _Float16* Kb = Kh + (size_t)bh * SEQ * HD;
  const _Float16* Vb = Vth + (size_t)bh * HD * SEQ;
  _Float16* pb = &pbuf[wid][0];

  // Q A-fragments for hd chunks [0,32) and [32,64) — loaded once.
  const v16h qa0 = load_a_frag(Qb + (size_t)q0 * HD + 0, HD, lane);
  const v16h qa1 = load_a_frag(Qb + (size_t)q0 * HD + 32, HD, lane);

  // Stage kb=0: K tile is 4KB contiguous; V tile rows are strided SEQ.
#pragma unroll
  for (int c = tid; c < 256; c += 128)
    async_b128(&Ks[0][c * 8], Kb + (size_t)c * 8);
#pragma unroll
  for (int c = tid; c < 256; c += 128) {
    const int row = c >> 2, q = c & 3;
    async_b128(&Vs[0][row * 32 + q * 8], Vb + (size_t)row * SEQ + q * 8);
  }

  v8f o[4] = {};
  float mrow[8], lrow[8];
#pragma unroll
  for (int r = 0; r < 8; ++r) { mrow[r] = -1.0e30f; lrow[r] = 0.0f; }

  for (int kb = 0; kb < SEQ; kb += 32) {
    const int buf = (kb >> 5) & 1;
    wait_async0();
    __syncthreads();
    if (kb + 32 < SEQ) {
      const _Float16* kn = Kb + (size_t)(kb + 32) * HD;
#pragma unroll
      for (int c = tid; c < 256; c += 128)
        async_b128(&Ks[buf ^ 1][c * 8], kn + (size_t)c * 8);
#pragma unroll
      for (int c = tid; c < 256; c += 128) {
        const int row = c >> 2, q = c & 3;
        async_b128(&Vs[buf ^ 1][row * 32 + q * 8],
                   Vb + (size_t)row * SEQ + (kb + 32) + q * 8);
      }
    }

    // Scores: S[16q][32k] = Q(16x64) x K^T(64x32), two 16-key C tiles.
    const _Float16* kt = &Ks[buf][0];
    v8f s0 = {}, s1 = {};
    s0 = wmma_f16(qa0, load_b_frag(kt + 0, HD, lane), s0);
    s0 = wmma_f16(qa1, load_b_frag(kt + 32, HD, lane), s0);
    s1 = wmma_f16(qa0, load_b_frag(kt + 16 * HD + 0, HD, lane), s1);
    s1 = wmma_f16(qa1, load_b_frag(kt + 16 * HD + 32, HD, lane), s1);

    // Online softmax (log2 domain; scale folded into Q).
    float alpha[8];
#pragma unroll
    for (int r = 0; r < 8; ++r) {
      float mx = fmaxf(s0[r], s1[r]);
#pragma unroll
      for (int off = 1; off < 16; off <<= 1)
        mx = fmaxf(mx, __shfl_xor(mx, off, 32));
      const float mnew = fmaxf(mrow[r], mx);
      alpha[r] = EXP2F(mrow[r] - mnew);
      const float p0 = EXP2F(s0[r] - mnew);
      const float p1 = EXP2F(s1[r] - mnew);
      float rs = p0 + p1;
#pragma unroll
      for (int off = 1; off < 16; off <<= 1) rs += __shfl_xor(rs, off, 32);
      lrow[r] = lrow[r] * alpha[r] + rs;
      mrow[r] = mnew;
      // Stage P tile to per-wave LDS in [row][key] order (row = rh*8+r).
      const int row = rh * 8 + r;
      pb[row * 32 + col] = (_Float16)p0;
      pb[row * 32 + 16 + col] = (_Float16)p1;
    }
#pragma unroll
    for (int j = 0; j < 4; ++j)
#pragma unroll
      for (int r = 0; r < 8; ++r) o[j][r] *= alpha[r];

    // Reload P as an A fragment (row = lane&15, keys contiguous).
    // Same-wave DS ops are in-order; compiler inserts s_wait_dscnt.
    const _Float16* pp = pb + (size_t)col * 32 + (rh << 3);
    v8h plo = *(const v8h*)pp;
    v8h phi = *(const v8h*)(pp + 16);
    v16h pa;
#pragma unroll
    for (int i = 0; i < 8; ++i) { pa[i] = plo[i]; pa[i + 8] = phi[i]; }

    // O += P(16x32) x V(32x64); V^T tile in LDS, B-frags contiguous.
#pragma unroll
    for (int j = 0; j < 4; ++j) {
      v16h vf = load_b_frag(&Vs[buf][j * 16 * 32], 32, lane);
      o[j] = wmma_f16(pa, vf, o[j]);
    }
    __syncthreads();  // done reading Ks/Vs[buf] before it is refilled
  }

  // Normalize and write O in [B, S, D] f16 for the projection GEMM.
  const int b = bh >> 4, head = bh & 15;
#pragma unroll
  for (int r = 0; r < 8; ++r) {
    const float inv = 1.0f / lrow[r];
    const int s = q0 + rh * 8 + r;
    const size_t rowoff = ((size_t)(b * SEQ + s)) * EMBED + head * HD + col;
#pragma unroll
    for (int j = 0; j < 4; ++j)
      Oh[rowoff + j * 16] = (_Float16)(o[j][r] * inv);
  }
}

// ---------------------------------------------------------------------------
// Kernel 3: output projection. out[m][n] = sum_k O[m][k]*Wp[n][k] + bp[n].
// Same LDS-staged structure as qkv_gemm, f32 epilogue.
// grid: (1024/64, 4096/64), block 128.
// ---------------------------------------------------------------------------
__global__ __launch_bounds__(128) void proj_gemm(
    const _Float16* __restrict__ Oh, const _Float16* __restrict__ wh,
    const float* __restrict__ bias, float* __restrict__ out) {
  __shared__ __align__(32) _Float16 Bs[2][64 * 32];
  const int tid = threadIdx.x;
  const int lane = tid & 31;
  const int wid = tid >> 5;
  const int m0 = blockIdx.y * 64 + wid * 16;
  const int n0 = blockIdx.x * 64;
  const _Float16* wpan = wh + (size_t)n0 * EMBED;

  stage_b_panel(wpan, Bs[0], tid);

  v8f acc[4] = {};
  for (int k = 0; k < EMBED; k += 32) {
    const int buf = (k >> 5) & 1;
    wait_async0();
    __syncthreads();
    if (k + 32 < EMBED) stage_b_panel(wpan + (k + 32), Bs[buf ^ 1], tid);

    v16h a = load_a_frag(Oh + (size_t)m0 * EMBED + k, EMBED, lane);
#pragma unroll
    for (int j = 0; j < 4; ++j) {
      v16h b = load_b_frag(&Bs[buf][j * 16 * 32], 32, lane);
      acc[j] = wmma_f16(a, b, acc[j]);
    }
    __syncthreads();
  }

  const int col = lane & 15;
  const int rh = lane >> 4;
#pragma unroll
  for (int j = 0; j < 4; ++j) {
    const int n = n0 + j * 16 + col;
    const float bv = bias[n];
#pragma unroll
    for (int r = 0; r < 8; ++r)
      out[(size_t)(m0 + rh * 8 + r) * EMBED + n] = acc[j][r] + bv;
  }
}

// ---------------------------------------------------------------------------
// Launch. Workspace (48 MB):
//   [ 0MB) xh      f16  4096x1024
//   [ 8MB) wqkvh   f16  3072x1024
//   [14MB) wprojh  f16  1024x1024
//   [16MB) Qh      f16  [32][2048][64] (pre-scaled)
//   [24MB) Kh      f16  [32][2048][64]
//   [32MB) Vth     f16  [32][64][2048] (transposed)
//   [40MB) Oh      f16  [4096][1024]
// ---------------------------------------------------------------------------
extern "C" void kernel_launch(void* const* d_in, const int* in_sizes, int n_in,
                              void* d_out, int out_size, void* d_ws,
                              size_t ws_size, hipStream_t stream) {
  const float* x = (const float*)d_in[0];
  const float* w_qkv = (const float*)d_in[1];
  const float* b_qkv = (const float*)d_in[2];
  const float* w_proj = (const float*)d_in[3];
  const float* b_proj = (const float*)d_in[4];
  float* out = (float*)d_out;

  char* ws = (char*)d_ws;
  _Float16* xh = (_Float16*)(ws + (size_t)0);
  _Float16* wqkvh = (_Float16*)(ws + ((size_t)8 << 20));
  _Float16* wprojh = (_Float16*)(ws + ((size_t)14 << 20));
  _Float16* Qh = (_Float16*)(ws + ((size_t)16 << 20));
  _Float16* Kh = (_Float16*)(ws + ((size_t)24 << 20));
  _Float16* Vth = (_Float16*)(ws + ((size_t)32 << 20));
  _Float16* Oh = (_Float16*)(ws + ((size_t)40 << 20));

  const int nx = MROWS * EMBED;        // 4,194,304
  const int nwq = 3 * EMBED * EMBED;   // 3,145,728
  const int nwp = EMBED * EMBED;       // 1,048,576

  cvt_f32_f16_x4<<<(nx / 4 + 255) / 256, 256, 0, stream>>>(
      (const v4f*)x, (v4h*)xh, nx / 4);
  cvt_f32_f16_x4<<<(nwq / 4 + 255) / 256, 256, 0, stream>>>(
      (const v4f*)w_qkv, (v4h*)wqkvh, nwq / 4);
  cvt_f32_f16_x4<<<(nwp / 4 + 255) / 256, 256, 0, stream>>>(
      (const v4f*)w_proj, (v4h*)wprojh, nwp / 4);

  qkv_gemm<<<dim3(3 * EMBED / 64, MROWS / 64), 128, 0, stream>>>(
      xh, wqkvh, b_qkv, Qh, Kh, Vth);

  flash_attn<<<dim3(SEQ / 64, BATCH * HEADS), 128, 0, stream>>>(
      Qh, Kh, Vth, Oh);

  proj_gemm<<<dim3(EMBED / 64, MROWS / 64), 128, 0, stream>>>(
      Oh, wprojh, b_proj, out);
}
